// GINE_5282809774138
// MI455X (gfx1250) — matrix operand order repaired and checked by reading, not compiled, not run on hardware
//
#include <hip/hip_runtime.h>
#include <hip/hip_bf16.h>

#define NN 100000      // nodes
#define NE 1600000     // edges
#define EMB 64
#define HID 128
#define NG 512         // graphs
#define NT 128         // tasks
#define BV 16          // bond vocab

typedef __attribute__((ext_vector_type(16))) __bf16 v16bf;
typedef __attribute__((ext_vector_type(8)))  __bf16 v8bf;
typedef __attribute__((ext_vector_type(8)))  float  v8f;

static __device__ __forceinline__ __bf16 f2bf(float f) {
  union { float f; unsigned u; } v; v.f = f;
  unsigned u = v.u;
  unsigned r = (u + 0x7FFFu + ((u >> 16) & 1u)) >> 16;   // round-to-nearest-even
  unsigned short h = (unsigned short)r;
  return __builtin_bit_cast(__bf16, h);
}

// proj[b][c] = sum_k bond_emb[b][k] * W[k][c] + bias[c]   (16 x Fout, tiny)
__global__ void k_bond_proj(const float* __restrict__ bond_emb,
                            const float* __restrict__ W,
                            const float* __restrict__ bias,
                            float* __restrict__ out, int Fout) {
  int t = blockIdx.x * blockDim.x + threadIdx.x;
  if (t >= BV * Fout) return;
  int b = t / Fout, c = t % Fout;
  float acc = bias[c];
  #pragma unroll 8
  for (int k = 0; k < EMB; ++k) acc += bond_emb[b * EMB + k] * W[k * Fout + c];
  out[t] = acc;
}

// Bt[n][k] = bf16(W[k][n])  — transpose weights to [out,in] bf16 for WMMA B frags
__global__ void k_wt_bf16(const float* __restrict__ W, __bf16* __restrict__ Bt,
                          int K, int Nc) {
  int t = blockIdx.x * blockDim.x + threadIdx.x;
  if (t >= K * Nc) return;
  int n = t / K, k = t % K;
  Bt[t] = f2bf(W[(size_t)k * Nc + n]);
}

// d[i] = bf16(s[i]), 8 elements per thread (memory-bound conversion pre-pass)
__global__ void k_f2bf_arr(const float* __restrict__ s, __bf16* __restrict__ d,
                           long long n8) {
  long long t = (long long)blockIdx.x * blockDim.x + threadIdx.x;
  if (t >= n8) return;
  const float4* sp = (const float4*)(s + t * 8);
  float4 a = sp[0], b = sp[1];
  v8bf o;
  o[0] = f2bf(a.x); o[1] = f2bf(a.y); o[2] = f2bf(a.z); o[3] = f2bf(a.w);
  o[4] = f2bf(b.x); o[5] = f2bf(b.y); o[6] = f2bf(b.z); o[7] = f2bf(b.w);
  *(v8bf*)(d + t * 8) = o;
}

// x0[node] = atom_emb[x_idx[node]]   (float4 per 16 lanes/node)
__global__ void k_gather(const int* __restrict__ idx, const float* __restrict__ emb,
                         float* __restrict__ x) {
  long long tid = (long long)blockIdx.x * blockDim.x + threadIdx.x;
  int node = (int)(tid >> 4), q = (int)(tid & 15);
  if (node >= NN) return;
  float4 v = *(const float4*)(emb + (size_t)idx[node] * EMB + q * 4);
  *(float4*)(x + (size_t)node * EMB + q * 4) = v;
}

__global__ void k_copy4(const float* __restrict__ src, float* __restrict__ dst, long long n4) {
  long long t = (long long)blockIdx.x * blockDim.x + threadIdx.x;
  if (t < n4) ((float4*)dst)[t] = ((const float4*)src)[t];
}

__global__ void k_zero(float* __restrict__ p, int n) {
  int t = blockIdx.x * blockDim.x + threadIdx.x;
  if (t < n) p[t] = 0.f;
}

// hacc[dst] += relu(x[src] + proj[ea])  — F/4 lanes per edge, float atomics (hit L2)
template <int F>
__global__ void k_edge(const float* __restrict__ x, const float* __restrict__ proj,
                       const int* __restrict__ src, const int* __restrict__ dst,
                       const int* __restrict__ ea, float* __restrict__ hacc) {
  const int FQ = F / 4;
  long long tid = (long long)blockIdx.x * blockDim.x + threadIdx.x;
  int e = (int)(tid / FQ), q = (int)(tid & (FQ - 1));
  if (e >= NE) return;
  int s = src[e], d = dst[e], a = ea[e];
  float4 xv = *(const float4*)(x + (size_t)s * F + q * 4);
  float4 pv = *(const float4*)(proj + (size_t)a * F + q * 4);
  float4 m;
  m.x = fmaxf(xv.x + pv.x, 0.f);
  m.y = fmaxf(xv.y + pv.y, 0.f);
  m.z = fmaxf(xv.z + pv.z, 0.f);
  m.w = fmaxf(xv.w + pv.w, 0.f);
  float* o = hacc + (size_t)d * F + q * 4;
  atomicAdd(o + 0, m.x); atomicAdd(o + 1, m.y);
  atomicAdd(o + 2, m.z); atomicAdd(o + 3, m.w);
}

// C[M,128] = act(A[M,K] @ Bt[128,K]^T + bias).  A is bf16 [M,K].
// 256 thr = 8 waves; wave w owns n-tile w; block owns 32 rows (two independent
// accumulators per wave -> back-to-back independent WMMAs, B frag reused).
// Inner loop is pure b128 loads + v_wmma (no conversion VALU).
// ACT: 0=none, 1=relu, 2=elu.  OUTBF: 1 -> store bf16, 0 -> store f32.
// Requires M % 32 == 0.
template <int K, int ACT, int OUTBF>
__global__ __launch_bounds__(256) void k_gemm_wmma(
    const __bf16* __restrict__ A, const __bf16* __restrict__ Bt,
    const float* __restrict__ bias, void* __restrict__ Cout) {
  const int lane = threadIdx.x & 31;
  const int wave = threadIdx.x >> 5;            // 0..7 -> n tile
  const int half = lane >> 4;                   // lane group 0/1
  const int m0 = blockIdx.x * 32;
  const int ncol = wave * 16 + (lane & 15);
  const __bf16* arow0 = A + (size_t)(m0 + (lane & 15)) * K + (half << 4);  // A: K base 0/16
  const __bf16* arow1 = arow0 + (size_t)16 * K;
  const __bf16* brow  = Bt + (size_t)ncol * K + (half << 4);               // B: K base 0/16
  v8f acc0 = {}, acc1 = {};
  #pragma unroll
  for (int kc = 0; kc < K; kc += 32) {
    v16bf bv  = *(const v16bf*)(brow + kc);
    v8bf a0l = *(const v8bf*)(arow0 + kc);       // K = base+0..7
    v8bf a0h = *(const v8bf*)(arow0 + kc + 8);   // K = base+8..15
    v8bf a1l = *(const v8bf*)(arow1 + kc);
    v8bf a1h = *(const v8bf*)(arow1 + kc + 8);
    v16bf av0 = __builtin_shufflevector(a0l, a0h, 0, 1, 2, 3, 4, 5, 6, 7,
                                        8, 9, 10, 11, 12, 13, 14, 15);
    v16bf av1 = __builtin_shufflevector(a1l, a1h, 0, 1, 2, 3, 4, 5, 6, 7,
                                        8, 9, 10, 11, 12, 13, 14, 15);
    acc0 = __builtin_amdgcn_wmma_f32_16x16x32_bf16(false, av0, false, bv,
                                                   (short)0, acc0, false, false);
    acc1 = __builtin_amdgcn_wmma_f32_16x16x32_bf16(false, av1, false, bv,
                                                   (short)0, acc1, false, false);
  }
  const float bb = bias[ncol];
  #pragma unroll
  for (int r = 0; r < 8; ++r) {
    int m = m0 + r + (half << 3);               // C/D layout: VGPR r -> row r (+8)
    float v0 = acc0[r] + bb;
    float v1 = acc1[r] + bb;
    if (ACT == 1) {
      v0 = fmaxf(v0, 0.f);
      v1 = fmaxf(v1, 0.f);
    } else if (ACT == 2) {
      v0 = (v0 > 0.f) ? v0 : (__expf(v0) - 1.f);
      v1 = (v1 > 0.f) ? v1 : (__expf(v1) - 1.f);
    }
    if (OUTBF) {
      __bf16* C = (__bf16*)Cout;
      C[(size_t)m * HID + ncol]        = f2bf(v0);
      C[(size_t)(m + 16) * HID + ncol] = f2bf(v1);
    } else {
      float* C = (float*)Cout;
      C[(size_t)m * HID + ncol]        = v0;
      C[(size_t)(m + 16) * HID + ncol] = v1;
    }
  }
}

// hg[batch[n]] += x[n]  (32 lanes/node, float4 each)
__global__ void k_pool(const float* __restrict__ x, const int* __restrict__ batch,
                       float* __restrict__ hg) {
  long long tid = (long long)blockIdx.x * blockDim.x + threadIdx.x;
  int node = (int)(tid >> 5), q = (int)(tid & 31);
  if (node >= NN) return;
  int g = batch[node];
  float4 v = *(const float4*)(x + (size_t)node * HID + q * 4);
  float* o = hg + (size_t)g * HID + q * 4;
  atomicAdd(o + 0, v.x); atomicAdd(o + 1, v.y);
  atomicAdd(o + 2, v.z); atomicAdd(o + 3, v.w);
}

extern "C" void kernel_launch(void* const* d_in, const int* in_sizes, int n_in,
                              void* d_out, int out_size, void* d_ws, size_t ws_size,
                              hipStream_t stream) {
  const int*   x_idx    = (const int*)d_in[0];
  const int*   eidx     = (const int*)d_in[1];
  const int*   src      = eidx;
  const int*   dst      = eidx + NE;
  const int*   ea       = (const int*)d_in[2];
  const int*   batch    = (const int*)d_in[3];
  const float* atom_emb = (const float*)d_in[4];
  const float* bond_emb = (const float*)d_in[5];
  const float* w_edge0  = (const float*)d_in[6];
  const float* b_edge0  = (const float*)d_in[7];
  const float* w1_0     = (const float*)d_in[8];
  const float* b1_0     = (const float*)d_in[9];
  const float* w2_0     = (const float*)d_in[10];
  const float* b2_0     = (const float*)d_in[11];
  const float* w_edge1  = (const float*)d_in[12];
  const float* b_edge1  = (const float*)d_in[13];
  const float* w1_1     = (const float*)d_in[14];
  const float* b1_1     = (const float*)d_in[15];
  const float* w2_1     = (const float*)d_in[16];
  const float* b2_1     = (const float*)d_in[17];
  const float* lin1_w   = (const float*)d_in[18];
  const float* lin1_b   = (const float*)d_in[19];
  const float* lin2_w   = (const float*)d_in[20];
  const float* lin2_b   = (const float*)d_in[21];
  float* out = (float*)d_out;

  // ---- workspace layout ----
  // f32 region: NN*384 floats
  float* W0 = (float*)d_ws;
  float* x0 = W0;                           // [NN,64]
  float* h0 = W0 + (size_t)NN * 64;         // [NN,64]   (= x0 + agg0, atomics)
  float* x1 = W0 + (size_t)NN * 128;        // [NN,128]
  float* h1 = W0 + (size_t)NN * 256;        // [NN,128]  (atomics)
  float* x2 = W0;                           // [NN,128]  reuse dead x0+h0
  // bf16 region: NN*320 bf16
  __bf16* BF   = (__bf16*)(W0 + (size_t)NN * 384);
  __bf16* h0bf = BF;                        // [NN,64]
  __bf16* m0bf = BF + (size_t)NN * 64;      // [NN,128]  (reused as m1)
  __bf16* h1bf = BF + (size_t)NN * 192;     // [NN,128]
  // small tail
  float* tailf = (float*)(BF + (size_t)NN * 320);
  float* proj0 = tailf;                     // 16*64
  float* proj1 = proj0 + 16 * 64;           // 16*128
  float* hg    = proj1 + 16 * 128;          // 512*128 (atomics)
  __bf16* hgbf = (__bf16*)(hg + NG * HID);  // 512*128
  __bf16* hrbf = hgbf + NG * HID;           // 512*128
  __bf16* w1_0t = hrbf + NG * HID;          // [128,64]
  __bf16* w2_0t = w1_0t + 128 * 64;         // [128,128]
  __bf16* w1_1t = w2_0t + 128 * 128;
  __bf16* w2_1t = w1_1t + 128 * 128;
  __bf16* lin1t = w2_1t + 128 * 128;
  __bf16* lin2t = lin1t + 128 * 128;

  // ---- tiny precomputes: bond projections + bf16 weight transposes ----
  k_bond_proj<<<(16 * 64 + 255) / 256, 256, 0, stream>>>(bond_emb, w_edge0, b_edge0, proj0, 64);
  k_bond_proj<<<(16 * 128 + 255) / 256, 256, 0, stream>>>(bond_emb, w_edge1, b_edge1, proj1, 128);
  k_wt_bf16<<<(64 * 128 + 255) / 256, 256, 0, stream>>>(w1_0, w1_0t, 64, 128);
  k_wt_bf16<<<(128 * 128 + 255) / 256, 256, 0, stream>>>(w2_0, w2_0t, 128, 128);
  k_wt_bf16<<<(128 * 128 + 255) / 256, 256, 0, stream>>>(w1_1, w1_1t, 128, 128);
  k_wt_bf16<<<(128 * 128 + 255) / 256, 256, 0, stream>>>(w2_1, w2_1t, 128, 128);
  k_wt_bf16<<<(128 * 128 + 255) / 256, 256, 0, stream>>>(lin1_w, lin1t, 128, 128);
  k_wt_bf16<<<(128 * 128 + 255) / 256, 256, 0, stream>>>(lin2_w, lin2t, 128, 128);

  // ---- layer 0 ----
  k_gather<<<((size_t)NN * 16 + 255) / 256, 256, 0, stream>>>(x_idx, atom_emb, x0);
  k_copy4<<<((size_t)NN * 16 + 255) / 256, 256, 0, stream>>>(x0, h0, (long long)NN * 16);
  k_edge<64><<<((long long)NE * 16 + 255) / 256, 256, 0, stream>>>(x0, proj0, src, dst, ea, h0);
  k_f2bf_arr<<<((size_t)NN * 8 + 255) / 256, 256, 0, stream>>>(h0, h0bf, (long long)NN * 8);
  k_gemm_wmma<64, 2, 1><<<NN / 32, 256, 0, stream>>>(h0bf, w1_0t, b1_0, m0bf);   // elu -> bf16
  k_gemm_wmma<128, 2, 0><<<NN / 32, 256, 0, stream>>>(m0bf, w2_0t, b2_0, x1);    // elu -> f32

  // ---- layer 1 ----
  k_copy4<<<((size_t)NN * 32 + 255) / 256, 256, 0, stream>>>(x1, h1, (long long)NN * 32);
  k_edge<128><<<((long long)NE * 32 + 255) / 256, 256, 0, stream>>>(x1, proj1, src, dst, ea, h1);
  k_f2bf_arr<<<((size_t)NN * 16 + 255) / 256, 256, 0, stream>>>(h1, h1bf, (long long)NN * 16);
  k_gemm_wmma<128, 2, 1><<<NN / 32, 256, 0, stream>>>(h1bf, w1_1t, b1_1, m0bf);  // m1, elu -> bf16
  k_gemm_wmma<128, 2, 0><<<NN / 32, 256, 0, stream>>>(m0bf, w2_1t, b2_1, x2);    // elu -> f32

  // ---- pool + head ----
  k_zero<<<(NG * HID + 255) / 256, 256, 0, stream>>>(hg, NG * HID);
  k_pool<<<((size_t)NN * 32 + 255) / 256, 256, 0, stream>>>(x2, batch, hg);
  k_f2bf_arr<<<(NG * HID / 8 + 255) / 256, 256, 0, stream>>>(hg, hgbf, NG * HID / 8);
  k_gemm_wmma<128, 1, 1><<<NG / 32, 256, 0, stream>>>(hgbf, lin1t, lin1_b, hrbf);  // relu -> bf16
  k_gemm_wmma<128, 0, 0><<<NG / 32, 256, 0, stream>>>(hrbf, lin2t, lin2_b, out);   // linear -> f32
}
